// Transformer_67181878444397
// MI455X (gfx1250) — matrix operand order repaired and checked by reading, not compiled, not run on hardware
//
#include <hip/hip_runtime.h>
#include <hip/hip_bf16.h>
#include <math.h>

// ---------------- model dims ----------------
static constexpr int CB = 2, CNC = 256, CNT = 512, CD = 128, CDFF = 512;
static constexpr int CH = 4, CEC = 512, CET = 1024, CELEM = 50, CNB = 8;

typedef __attribute__((ext_vector_type(16))) _Float16     v16h;
typedef __attribute__((ext_vector_type(8)))  float        v8f;
typedef __attribute__((ext_vector_type(4)))  float        f4;
typedef __attribute__((ext_vector_type(4)))  unsigned int u4;

// ============================================================
// Generalized batched strided GEMM:  C = relu?(alpha * A*B + bias)
// A: f32 or f16 (template AF16), elem(i,k) at offA + i*rsA + k*csA
// B: f32,                        elem(k,n) at offB + k*rsB + n*csB
// C: f32 or f16 (template CF16), elem(i,n) at offC + i*rsC + n*csC
// per-z batch offsets: off = (z/div)*sHi + (z%div)*sLo
// ============================================================
struct GemmParams {
  const void*  A;
  const float* Bm;
  const float* bias;
  void*        C;
  long long rsA, csA, rsB, csB, rsC, csC;
  long long sAHi, sALo, sBHi, sBLo, sBiasHi, sBiasLo, sCHi, sCLo;
  int M, N, K;
  int divA, divB, divBias, divC;
  float alpha;
  int aF16, cF16, relu;
};

template <int AF16, int CF16>
__global__ __launch_bounds__(128) void gemm_wmma(GemmParams p)
{
  __shared__ alignas(16) _Float16 As[64][32];   // [m][k]
  __shared__ alignas(16) _Float16 Bs[64][32];   // [n][k]
  const int z = blockIdx.z;
  const long long offA    = (long long)(z / p.divA)    * p.sAHi    + (long long)(z % p.divA)    * p.sALo;
  const long long offB    = (long long)(z / p.divB)    * p.sBHi    + (long long)(z % p.divB)    * p.sBLo;
  const long long offBias = (long long)(z / p.divBias) * p.sBiasHi + (long long)(z % p.divBias) * p.sBiasLo;
  const long long offC    = (long long)(z / p.divC)    * p.sCHi    + (long long)(z % p.divC)    * p.sCLo;
  const int m0 = blockIdx.y << 6;
  const int n0 = blockIdx.x << 6;
  const int tid = threadIdx.x;
  const int wave = tid >> 5;
  const int lane = tid & 31;
  const float*    Af = (const float*)p.A;
  const _Float16* Ah = (const _Float16*)p.A;

  // block-uniform fast-path eligibility (alignment + full-tile checks)
  const int alignReq = AF16 ? 7 : 3;
  const bool aVec  = (p.csA == 1) && ((m0 + 64) <= p.M) &&
                     ((p.rsA & alignReq) == 0) && ((offA & alignReq) == 0);
  const bool bVecK = (p.rsB == 1) && ((n0 + 64) <= p.N) &&
                     ((p.csB & 3) == 0) && ((offB & 3) == 0);
  const bool bVecN = (p.csB == 1) && ((n0 + 64) <= p.N) &&
                     ((p.rsB & 3) == 0) && ((offB & 3) == 0);

  v8f acc[4] = {};

  for (int k0 = 0; k0 < p.K; k0 += 32) {
    const bool kFull = (k0 + 32) <= p.K;

    // ------------- stage A tile (64x32, f16) -------------
    if (aVec && kFull) {
      int row = tid >> 1, kh = (tid & 1) << 4;           // 16 contiguous k per thread
      long long base = offA + (long long)(m0 + row) * p.rsA + (k0 + kh);
      unsigned int* dst = (unsigned int*)(&As[row][0]) + (kh >> 1);
      if (AF16) {
        const u4* src = (const u4*)(Ah + base);
        u4 a0 = src[0], a1 = src[1];
        #pragma unroll
        for (int j = 0; j < 4; ++j) { dst[j] = a0[j]; dst[4 + j] = a1[j]; }
        if (k0 + 32 < p.K) __builtin_prefetch((const void*)(Ah + base + 32), 0, 1);
      } else {
        union { _Float16 h[16]; unsigned int u[8]; } t;
        #pragma unroll
        for (int i = 0; i < 4; ++i) {
          f4 vv = *(const f4*)(Af + base + 4 * i);
          #pragma unroll
          for (int c = 0; c < 4; ++c) t.h[4 * i + c] = (_Float16)vv[c];
        }
        #pragma unroll
        for (int j = 0; j < 8; ++j) dst[j] = t.u[j];
        if (k0 + 32 < p.K) __builtin_prefetch((const void*)(Af + base + 32), 0, 1);
      }
    } else {
      #pragma unroll
      for (int r = 0; r < 4; ++r) {
        int idx = tid + (r << 7) * 4;       // unrolled differently below
        (void)idx;
      }
      for (int r = 0; r < 16; ++r) {
        int idx = tid + (r << 7);
        int mi = idx >> 5, ki = idx & 31;
        float av = 0.0f;
        if ((m0 + mi) < p.M && (k0 + ki) < p.K) {
          long long ao = offA + (long long)(m0 + mi) * p.rsA + (long long)(k0 + ki) * p.csA;
          av = AF16 ? (float)Ah[ao] : Af[ao];
        }
        As[mi][ki] = (_Float16)av;
      }
    }

    // ------------- stage B tile (n-major 64x32, f16) -------------
    if (bVecK && kFull) {
      int n = tid >> 1, kh = (tid & 1) << 4;             // 16 contiguous k per thread
      long long base = offB + (long long)(n0 + n) * p.csB + (k0 + kh);
      unsigned int* dst = (unsigned int*)(&Bs[n][0]) + (kh >> 1);
      union { _Float16 h[16]; unsigned int u[8]; } t;
      #pragma unroll
      for (int i = 0; i < 4; ++i) {
        f4 vv = *(const f4*)(p.Bm + base + 4 * i);
        #pragma unroll
        for (int c = 0; c < 4; ++c) t.h[4 * i + c] = (_Float16)vv[c];
      }
      #pragma unroll
      for (int j = 0; j < 8; ++j) dst[j] = t.u[j];
      if (k0 + 32 < p.K) __builtin_prefetch((const void*)(p.Bm + base + 32), 0, 1);
    } else if (bVecN && kFull) {
      int k = tid & 31, nq = (tid >> 5) << 4;            // 16 contiguous n per thread
      long long base = offB + (long long)(k0 + k) * p.rsB + (n0 + nq);
      #pragma unroll
      for (int i = 0; i < 4; ++i) {
        f4 vv = *(const f4*)(p.Bm + base + 4 * i);
        #pragma unroll
        for (int c = 0; c < 4; ++c) Bs[nq + 4 * i + c][k] = (_Float16)vv[c];
      }
      if (k0 + 32 < p.K) __builtin_prefetch((const void*)(p.Bm + base + 32 * p.rsB), 0, 1);
    } else {
      for (int r = 0; r < 16; ++r) {
        int idx = tid + (r << 7);
        int ni = idx >> 5, ki = idx & 31;
        float bv = 0.0f;
        if ((n0 + ni) < p.N && (k0 + ki) < p.K) {
          long long bo = offB + (long long)(k0 + ki) * p.rsB + (long long)(n0 + ni) * p.csB;
          bv = p.Bm[bo];
        }
        Bs[ni][ki] = (_Float16)bv;
      }
    }
    __syncthreads();

    // A fragment: contiguous k order, same as B (lanes 0-15: K 0..15,
    // lanes 16-31: K 16..31, packed pairs)
    union { unsigned int u[8]; v16h h; } afr, bfr;
    const unsigned int* ar = (const unsigned int*)(&As[(wave << 4) + (lane & 15)][0]);
    const int ka = (lane < 16) ? 0 : 8;   // dword base within row
    #pragma unroll
    for (int j = 0; j < 4; ++j) { afr.u[j] = ar[ka + j]; afr.u[4 + j] = ar[ka + 4 + j]; }
    const int kb = (lane < 16) ? 0 : 8;   // B: lanes 0-15 K 0..15, lanes 16-31 K 16..31
    #pragma unroll
    for (int nt = 0; nt < 4; ++nt) {
      const unsigned int* br = (const unsigned int*)(&Bs[(nt << 4) + (lane & 15)][0]);
      #pragma unroll
      for (int j = 0; j < 8; ++j) bfr.u[j] = br[kb + j];
      acc[nt] = __builtin_amdgcn_wmma_f32_16x16x32_f16(
          false, afr.h, false, bfr.h, (short)0, acc[nt], false, false);
    }
    __syncthreads();
  }

  // epilogue: C/D layout, lane holds rows (lane>=16?8:0)+j, col = lane&15
  const int rowOff = (lane >= 16) ? 8 : 0;
  #pragma unroll
  for (int nt = 0; nt < 4; ++nt) {
    int ncol = n0 + (nt << 4) + (lane & 15);
    if (ncol >= p.N) continue;
    float bval = p.bias ? p.bias[offBias + ncol] : 0.0f;
    #pragma unroll
    for (int j = 0; j < 8; ++j) {
      int mrow = m0 + (wave << 4) + rowOff + j;
      if (mrow >= p.M) continue;
      float v = acc[nt][j] * p.alpha + bval;
      if (p.relu && v < 0.0f) v = 0.0f;
      long long co = offC + (long long)mrow * p.rsC + (long long)ncol * p.csC;
      if (CF16) ((_Float16*)p.C)[co] = (_Float16)v;
      else      ((float*)p.C)[co] = v;
    }
  }
}

// ============================================================
// small helper kernels
// ============================================================
__global__ void k_zero(float* p, long long n)
{
  long long i = (long long)blockIdx.x * 256 + threadIdx.x;
  if (i < n) p[i] = 0.0f;
}

__global__ void k_compound_concat(const int* E, const int* C, const float* elemEmb,
                                  const float* chargeEmb, float* out, int rows)
{
  int row = blockIdx.x, t = threadIdx.x;
  if (row >= rows) return;
  out[(long long)row * (3 * CD) + t]      = elemEmb[(long long)E[row] * CD + t];
  out[(long long)row * (3 * CD) + CD + t] = chargeEmb[(long long)C[row] * CD + t];
}

__global__ void k_theo_concat(const int* E, const float* elemEmb, float* out, int rows)
{
  int row = blockIdx.x, t = threadIdx.x;
  if (row >= rows) return;
  out[(long long)row * (2 * CD) + t] = elemEmb[(long long)E[row] * CD + t];
}

__global__ void k_bond_wsum(const int* BT, const float* bondEmb, float* wsum, int n)
{
  int i = blockIdx.x * 128 + threadIdx.x;
  if (i >= n) return;
  int b = BT[i];
  float s = 0.f;
  for (int d = 0; d < CD; ++d) s += bondEmb[(long long)b * CD + d];
  wsum[i] = s;
}

__global__ void k_theo_wsum(const int* TBT, float* wsumT, int perB)
{
  __shared__ float red[256];
  int b = blockIdx.x, tid = threadIdx.x;
  float s = 0.f;
  for (int e = tid; e < perB; e += 256) s += (float)TBT[(long long)b * perB + e];
  red[tid] = s; __syncthreads();
  for (int st = 128; st > 0; st >>= 1) { if (tid < st) red[tid] += red[tid + st]; __syncthreads(); }
  if (tid == 0) wsumT[b] = red[0];
}

__global__ void k_graph_scatter(const int* idx, const float* s, const float* wsum,
                                float* zb, int N, int nB, int scalarW, int total)
{
  int gi = blockIdx.x * 256 + threadIdx.x;
  if (gi >= total) return;
  int b = gi / nB, e = gi % nB;
  int i = idx[(long long)(b * 2) * nB + e];
  int j = idx[(long long)(b * 2 + 1) * nB + e];
  float wv = scalarW ? wsum[b] : wsum[(long long)b * nB + e];
  long long base = (((long long)b * N + i) * N + j) * CH;
  #pragma unroll
  for (int h = 0; h < CH; ++h) zb[base + h] = s[base + h] * wv;
}

// softmax over contiguous rows (b,h,n,:) — standard attention
__global__ void k_softmax_rows(float* s, int Mk, int Nq, int causal)
{
  __shared__ float red[256];
  long long row = blockIdx.x;
  int n = (int)(row % Nq);
  float* r = s + row * (long long)Mk;
  int tid = threadIdx.x;
  float mx = -3e38f;
  for (int m = tid; m < Mk; m += 256) {
    float v = (causal && m > n) ? -3e38f : r[m];
    mx = fmaxf(mx, v);
  }
  red[tid] = mx; __syncthreads();
  for (int st = 128; st > 0; st >>= 1) { if (tid < st) red[tid] = fmaxf(red[tid], red[tid + st]); __syncthreads(); }
  mx = red[0]; __syncthreads();
  float sm = 0.f;
  for (int m = tid; m < Mk; m += 256) {
    float v = (causal && m > n) ? 0.f : __expf(r[m] - mx);
    r[m] = v; sm += v;
  }
  red[tid] = sm; __syncthreads();
  for (int st = 128; st > 0; st >>= 1) { if (tid < st) red[tid] += red[tid + st]; __syncthreads(); }
  float inv = 1.0f / red[0];
  for (int m = tid; m < Mk; m += 256) r[m] *= inv;
}

// softmax over m of z (b,n,m,h), stride H
__global__ void k_softmax_graph(float* zb, int N, int causal)
{
  __shared__ float red[256];
  int blk = blockIdx.x;
  int h = blk % CH;
  int n = (blk / CH) % N;
  int b = blk / (CH * N);
  float* base = zb + (((long long)b * N + n) * (long long)N) * CH + h;
  int tid = threadIdx.x;
  float mx = -3e38f;
  for (int m = tid; m < N; m += 256) {
    float v = (causal && m > n) ? -3e38f : base[(long long)m * CH];
    mx = fmaxf(mx, v);
  }
  red[tid] = mx; __syncthreads();
  for (int st = 128; st > 0; st >>= 1) { if (tid < st) red[tid] = fmaxf(red[tid], red[tid + st]); __syncthreads(); }
  mx = red[0]; __syncthreads();
  float sm = 0.f;
  for (int m = tid; m < N; m += 256) {
    float v = (causal && m > n) ? 0.f : __expf(base[(long long)m * CH] - mx);
    base[(long long)m * CH] = v; sm += v;
  }
  red[tid] = sm; __syncthreads();
  for (int st = 128; st > 0; st >>= 1) { if (tid < st) red[tid] += red[tid + st]; __syncthreads(); }
  float inv = 1.0f / red[0];
  for (int m = tid; m < N; m += 256) base[(long long)m * CH] *= inv;
}

// softmax over axis 0 (B=2) of elem (B,NT,ELEM)
__global__ void k_softmax_axis0(float* e, int per)
{
  int i = blockIdx.x * 256 + threadIdx.x;
  if (i >= per) return;
  float a = e[i], b = e[per + i];
  float m = fmaxf(a, b);
  float ea = __expf(a - m), eb = __expf(b - m);
  float inv = 1.0f / (ea + eb);
  e[i] = ea * inv; e[per + i] = eb * inv;
}

// softmax over axis 1 (t1) of bonds (B,NT,NT,NB)
__global__ void k_softmax_bonds(float* bonds)
{
  __shared__ float red[256];
  int blk = blockIdx.x;
  int nb = blk % CNB;
  int t2 = (blk / CNB) % CNT;
  int b  = blk / (CNB * CNT);
  float* base = bonds + (long long)b * CNT * CNT * CNB + (long long)t2 * CNB + nb;
  const long long str = (long long)CNT * CNB;
  int tid = threadIdx.x;
  float mx = -3e38f;
  for (int t1 = tid; t1 < CNT; t1 += 256) mx = fmaxf(mx, base[t1 * str]);
  red[tid] = mx; __syncthreads();
  for (int st = 128; st > 0; st >>= 1) { if (tid < st) red[tid] = fmaxf(red[tid], red[tid + st]); __syncthreads(); }
  mx = red[0]; __syncthreads();
  float sm = 0.f;
  for (int t1 = tid; t1 < CNT; t1 += 256) { float v = __expf(base[t1 * str] - mx); base[t1 * str] = v; sm += v; }
  red[tid] = sm; __syncthreads();
  for (int st = 128; st > 0; st >>= 1) { if (tid < st) red[tid] += red[tid + st]; __syncthreads(); }
  float inv = 1.0f / red[0];
  for (int t1 = tid; t1 < CNT; t1 += 256) base[t1 * str] *= inv;
}

// out = LayerNorm(xin + res) * g + b   (D=128, one row per block)
__global__ void k_ln_res(const float* xin, const float* res,
                         const float* g, const float* bb, float* out)
{
  __shared__ float red[CD];
  long long row = blockIdx.x;
  int t = threadIdx.x;
  float v = xin[row * CD + t] + res[row * CD + t];
  red[t] = v; __syncthreads();
  for (int st = 64; st > 0; st >>= 1) { if (t < st) red[t] += red[t + st]; __syncthreads(); }
  float mu = red[0] * (1.0f / CD); __syncthreads();
  float d = v - mu;
  red[t] = d * d; __syncthreads();
  for (int st = 64; st > 0; st >>= 1) { if (t < st) red[t] += red[t + st]; __syncthreads(); }
  float var = red[0] * (1.0f / CD);
  out[row * CD + t] = d * rsqrtf(var + 1e-5f) * g[t] + bb[t];
}

// ============================================================
// host-side helpers
// ============================================================
struct Lin  { const float* b; const float* w; };
struct MhaW { const float *bk, *bq, *bv; Lin out; const float *wk, *wq, *wv; };
struct LnW  { const float* b; const float* g; };

struct Scratch {
  float *q, *k, *v, *s, *z, *o, *attnOut, *ffTmp, *tmpD;
  float *xcat, *tcat, *xcat2, *xA, *xB, *tA, *tB, *tC, *wsum, *wsumT;
  _Float16 *u16, *v16;
};

static inline GemmParams mkgp(const void* A, int aF16, const float* Bm, const float* bias,
                              void* C, int cF16, int M, int N, int K)
{
  GemmParams p;
  p.A = A; p.Bm = Bm; p.bias = bias; p.C = C;
  p.rsA = K; p.csA = 1; p.rsB = N; p.csB = 1; p.rsC = N; p.csC = 1;
  p.sAHi = 0; p.sALo = 0; p.sBHi = 0; p.sBLo = 0;
  p.sBiasHi = 0; p.sBiasLo = 0; p.sCHi = 0; p.sCLo = 0;
  p.M = M; p.N = N; p.K = K;
  p.divA = 1; p.divB = 1; p.divBias = 1; p.divC = 1;
  p.alpha = 1.0f; p.aF16 = aF16; p.cF16 = cF16; p.relu = 0;
  return p;
}

static inline void launch_gemm(const GemmParams& p, int batches, hipStream_t st)
{
  dim3 grid((p.N + 63) / 64, (p.M + 63) / 64, batches);
  dim3 blk(128);
  if (p.aF16) {
    if (p.cF16) gemm_wmma<1, 1><<<grid, blk, 0, st>>>(p);
    else        gemm_wmma<1, 0><<<grid, blk, 0, st>>>(p);
  } else {
    if (p.cF16) gemm_wmma<0, 1><<<grid, blk, 0, st>>>(p);
    else        gemm_wmma<0, 0><<<grid, blk, 0, st>>>(p);
  }
}

// standard multi-head attention; outbuf gets (B*Nq, D) rows with row stride outRs
static void run_mha(hipStream_t st, Scratch& S, const float* Qin, const float* KVin,
                    int Nq, int Mk, const MhaW& w, float* outbuf, long long outRs, int causal)
{
  const int BZ = CB * CH;
  { GemmParams p = mkgp(Qin, 0, w.wq, w.bq, S.q, 0, Nq, CD, CD);
    p.divA = CH; p.sAHi = (long long)Nq * CD;
    p.divB = CH; p.sBLo = (long long)CD * CD;
    p.divBias = CH; p.sBiasLo = CD;
    p.sCHi = (long long)Nq * CD;
    launch_gemm(p, BZ, st); }
  { GemmParams p = mkgp(KVin, 0, w.wk, w.bk, S.k, 0, Mk, CD, CD);
    p.divA = CH; p.sAHi = (long long)Mk * CD;
    p.divB = CH; p.sBLo = (long long)CD * CD;
    p.divBias = CH; p.sBiasLo = CD;
    p.sCHi = (long long)Mk * CD;
    launch_gemm(p, BZ, st); }
  { GemmParams p = mkgp(KVin, 0, w.wv, w.bv, S.v, 0, Mk, CD, CD);
    p.divA = CH; p.sAHi = (long long)Mk * CD;
    p.divB = CH; p.sBLo = (long long)CD * CD;
    p.divBias = CH; p.sBiasLo = CD;
    p.sCHi = (long long)Mk * CD;
    launch_gemm(p, BZ, st); }
  // scores: q @ k^T / sqrt(Mk)   -> s (z, Nq, Mk)
  { GemmParams p = mkgp(S.q, 0, S.k, nullptr, S.s, 0, Nq, Mk, CD);
    p.rsB = 1; p.csB = CD;
    p.sAHi = (long long)Nq * CD; p.sBHi = (long long)Mk * CD; p.sCHi = (long long)Nq * Mk;
    p.alpha = 1.0f / sqrtf((float)Mk);
    launch_gemm(p, BZ, st); }
  k_softmax_rows<<<dim3(BZ * Nq), dim3(256), 0, st>>>(S.s, Mk, Nq, causal);
  // o = a @ v  -> (b, n, h, d) packed for head-concat
  { GemmParams p = mkgp(S.s, 0, S.v, nullptr, S.o, 0, Nq, CD, Mk);
    p.sAHi = (long long)Nq * Mk; p.sBHi = (long long)Mk * CD;
    p.divC = CH; p.sCHi = (long long)Nq * CH * CD; p.sCLo = CD; p.rsC = CH * CD;
    launch_gemm(p, BZ, st); }
  { GemmParams p = mkgp(S.o, 0, w.out.w, w.out.b, outbuf, 0, CB * Nq, CD, CH * CD);
    p.rsC = outRs;
    launch_gemm(p, 1, st); }
}

// graph attention (scatter of scores at bond indices, softmax over m)
static void run_graph_mha(hipStream_t st, Scratch& S, const float* x, int N,
                          const int* idx, int nB, const float* wsum, int scalarW,
                          const MhaW& w, float* outbuf, long long outRs, int causal)
{
  const int BZ = CB * CH;
  { GemmParams p = mkgp(x, 0, w.wq, w.bq, S.q, 0, N, CD, CD);
    p.divA = CH; p.sAHi = (long long)N * CD;
    p.divB = CH; p.sBLo = (long long)CD * CD;
    p.divBias = CH; p.sBiasLo = CD;
    p.sCHi = (long long)N * CD;
    launch_gemm(p, BZ, st); }
  { GemmParams p = mkgp(x, 0, w.wk, w.bk, S.k, 0, N, CD, CD);
    p.divA = CH; p.sAHi = (long long)N * CD;
    p.divB = CH; p.sBLo = (long long)CD * CD;
    p.divBias = CH; p.sBiasLo = CD;
    p.sCHi = (long long)N * CD;
    launch_gemm(p, BZ, st); }
  { GemmParams p = mkgp(x, 0, w.wv, w.bv, S.v, 0, N, CD, CD);
    p.divA = CH; p.sAHi = (long long)N * CD;
    p.divB = CH; p.sBLo = (long long)CD * CD;
    p.divBias = CH; p.sBiasLo = CD;
    p.sCHi = (long long)N * CD;
    launch_gemm(p, BZ, st); }
  // scores stored (b,n,m,h)
  { GemmParams p = mkgp(S.q, 0, S.k, nullptr, S.s, 0, N, N, CD);
    p.rsB = 1; p.csB = CD;
    p.sAHi = (long long)N * CD; p.sBHi = (long long)N * CD;
    p.divC = CH; p.sCHi = (long long)N * N * CH; p.sCLo = 1;
    p.rsC = (long long)N * CH; p.csC = CH;
    p.alpha = 1.0f / sqrtf((float)N);
    launch_gemm(p, BZ, st); }
  long long ztot = (long long)CB * N * N * CH;
  k_zero<<<dim3((unsigned)((ztot + 255) / 256)), dim3(256), 0, st>>>(S.z, ztot);
  int tot = CB * nB;
  k_graph_scatter<<<dim3((tot + 255) / 256), dim3(256), 0, st>>>(idx, S.s, wsum, S.z, N, nB, scalarW, tot);
  k_softmax_graph<<<dim3(CB * N * CH), dim3(256), 0, st>>>(S.z, N, causal);
  // o = a @ v  with A = z (b,n,m,h)  (strided A -> scalar staging path)
  { GemmParams p = mkgp(S.z, 0, S.v, nullptr, S.o, 0, N, CD, N);
    p.divA = CH; p.sAHi = (long long)N * N * CH; p.sALo = 1;
    p.rsA = (long long)N * CH; p.csA = CH;
    p.sBHi = (long long)N * CD;
    p.divC = CH; p.sCHi = (long long)N * CH * CD; p.sCLo = CD; p.rsC = CH * CD;
    launch_gemm(p, BZ, st); }
  { GemmParams p = mkgp(S.o, 0, w.out.w, w.out.b, outbuf, 0, CB * N, CD, CH * CD);
    p.rsC = outRs;
    launch_gemm(p, 1, st); }
}

// ============================================================
// entry point
// ============================================================
extern "C" void kernel_launch(void* const* d_in, const int* in_sizes, int n_in,
                              void* d_out, int out_size, void* d_ws, size_t ws_size,
                              hipStream_t stream)
{
  (void)in_sizes; (void)n_in; (void)out_size; (void)ws_size;

  // raw inputs (setup_inputs dict order)
  const int*   RE  = (const int*)d_in[0];
  const int*   RC  = (const int*)d_in[1];
  const float* RP  = (const float*)d_in[2];
  const int*   RBI = (const int*)d_in[3];
  const int*   RBT = (const int*)d_in[4];
  const int*   PE  = (const int*)d_in[5];
  const int*   PC  = (const int*)d_in[6];
  const float* PP  = (const float*)d_in[7];
  const int*   PBI = (const int*)d_in[8];
  const int*   PBT = (const int*)d_in[9];
  const int*   TE  = (const int*)d_in[10];
  const float* TP  = (const float*)d_in[11];
  const int*   TBI = (const int*)d_in[12];
  const int*   TBT = (const int*)d_in[13];

  // params: tree_flatten order (dict keys sorted, recursively)
  int ip = 14;
  auto NF  = [&]() { return (const float*)d_in[ip++]; };
  auto GL  = [&]() { Lin l; l.b = NF(); l.w = NF(); return l; };
  auto GM  = [&]() { MhaW m; m.bk = NF(); m.bq = NF(); m.bv = NF();
                     m.out = GL(); m.wk = NF(); m.wq = NF(); m.wv = NF(); return m; };
  auto GLN = [&]() { LnW l; l.b = NF(); l.g = NF(); return l; };

  // top-level sorted: compound_ff, dec, enc, out, product, reactant, theozyme
  Lin compound_ff = GL();
  struct DecW { MhaW attn, cross; Lin ff1, ff2; LnW ln; } dec[4];
  for (int l = 0; l < 4; ++l) {
    dec[l].attn = GM(); dec[l].cross = GM();
    dec[l].ff1 = GL(); dec[l].ff2 = GL(); dec[l].ln = GLN();
  }
  struct EncW { Lin ff1, ff2; LnW ln; MhaW mha; } enc[4];
  for (int l = 0; l < 4; ++l) {
    enc[l].ff1 = GL(); enc[l].ff2 = GL(); enc[l].ln = GLN(); enc[l].mha = GM();
  }
  // out: b1, b2a, b2b, e1, e2, p1, p2
  Lin w_b1 = GL(), w_b2a = GL(), w_b2b = GL(), w_e1 = GL(), w_e2 = GL(), w_p1 = GL(), w_p2 = GL();
  // compound: bond_emb, charge_emb, elem_emb, ff, ga, pos
  struct CompW { const float *bond, *charge, *elem; Lin ff; MhaW ga; Lin pos; };
  CompW prod; prod.bond = NF(); prod.charge = NF(); prod.elem = NF();
  prod.ff = GL(); prod.ga = GM(); prod.pos = GL();
  CompW reac; reac.bond = NF(); reac.charge = NF(); reac.elem = NF();
  reac.ff = GL(); reac.ga = GM(); reac.pos = GL();
  // theozyme: elem_emb, ff, ga, pos
  struct TheoW { const float* elem; Lin ff; MhaW ga; Lin pos; };
  TheoW theo; theo.elem = NF(); theo.ff = GL(); theo.ga = GM(); theo.pos = GL();

  // ---------------- workspace carve ----------------
  char* wp = (char*)d_ws;
  auto carve = [&](size_t elems, size_t esz) -> void* {
    void* r = (void*)wp; wp += (elems * esz + 255) & ~(size_t)255; return r;
  };
  Scratch S;
  S.q       = (float*)carve((size_t)CB * CH * CNT * CD, 4);
  S.k       = (float*)carve((size_t)CB * CH * CNT * CD, 4);
  S.v       = (float*)carve((size_t)CB * CH * CNT * CD, 4);
  S.s       = (float*)carve((size_t)CB * CH * CNT * CNT, 4);
  S.z       = (float*)carve((size_t)CB * CH * CNT * CNT, 4);
  S.o       = (float*)carve((size_t)CB * CNT * CH * CD, 4);
  S.attnOut = (float*)carve((size_t)CB * CNT * CD, 4);
  S.ffTmp   = (float*)carve((size_t)CB * CNT * CDFF, 4);
  S.tmpD    = (float*)carve((size_t)CB * CNT * CD, 4);
  S.xcat    = (float*)carve((size_t)CB * CNC * 3 * CD, 4);
  S.tcat    = (float*)carve((size_t)CB * CNT * 2 * CD, 4);
  S.xcat2   = (float*)carve((size_t)CB * CNC * 2 * CD, 4);
  S.xA      = (float*)carve((size_t)CB * CNC * CD, 4);
  S.xB      = (float*)carve((size_t)CB * CNC * CD, 4);
  S.tA      = (float*)carve((size_t)CB * CNT * CD, 4);
  S.tB      = (float*)carve((size_t)CB * CNT * CD, 4);
  S.tC      = (float*)carve((size_t)CB * CNT * CD, 4);
  S.wsum    = (float*)carve((size_t)CB * CEC, 4);
  S.wsumT   = (float*)carve(16, 4);
  S.u16     = (_Float16*)carve((size_t)128 * CNT * CD, 2);  // 128-row slab of u
  S.v16     = (_Float16*)carve((size_t)128 * CNT * CD, 2);  // slab after b2a

  // ---------------- compound encoders ----------------
  // reactant -> xcat2[:, 0:128]
  k_compound_concat<<<CB * CNC, 128, 0, stream>>>(RE, RC, reac.elem, reac.charge, S.xcat, CB * CNC);
  { GemmParams p = mkgp(RP, 0, reac.pos.w, reac.pos.b, S.xcat + 2 * CD, 0, CB * CNC, CD, 3);
    p.rsC = 3 * CD; p.relu = 1; launch_gemm(p, 1, stream); }
  { GemmParams p = mkgp(S.xcat, 0, reac.ff.w, reac.ff.b, S.tmpD, 0, CB * CNC, CD, 3 * CD);
    p.relu = 1; launch_gemm(p, 1, stream); }
  k_bond_wsum<<<(CB * CEC + 127) / 128, 128, 0, stream>>>(RBT, reac.bond, S.wsum, CB * CEC);
  run_graph_mha(stream, S, S.tmpD, CNC, RBI, CEC, S.wsum, 0, reac.ga, S.xcat2, 2 * CD, 0);

  // product -> xcat2[:, 128:256]
  k_compound_concat<<<CB * CNC, 128, 0, stream>>>(PE, PC, prod.elem, prod.charge, S.xcat, CB * CNC);
  { GemmParams p = mkgp(PP, 0, prod.pos.w, prod.pos.b, S.xcat + 2 * CD, 0, CB * CNC, CD, 3);
    p.rsC = 3 * CD; p.relu = 1; launch_gemm(p, 1, stream); }
  { GemmParams p = mkgp(S.xcat, 0, prod.ff.w, prod.ff.b, S.tmpD, 0, CB * CNC, CD, 3 * CD);
    p.relu = 1; launch_gemm(p, 1, stream); }
  k_bond_wsum<<<(CB * CEC + 127) / 128, 128, 0, stream>>>(PBT, prod.bond, S.wsum, CB * CEC);
  run_graph_mha(stream, S, S.tmpD, CNC, PBI, CEC, S.wsum, 0, prod.ga, S.xcat2 + CD, 2 * CD, 0);

  // x = relu(linear(concat(rx,px), compound_ff))
  { GemmParams p = mkgp(S.xcat2, 0, compound_ff.w, compound_ff.b, S.xA, 0, CB * CNC, CD, 2 * CD);
    p.relu = 1; launch_gemm(p, 1, stream); }

  // ---------------- theozyme ----------------
  k_theo_concat<<<CB * CNT, 128, 0, stream>>>(TE, theo.elem, S.tcat, CB * CNT);
  { GemmParams p = mkgp(TP, 0, theo.pos.w, theo.pos.b, S.tcat + CD, 0, CB * CNT, CD, 3);
    p.rsC = 2 * CD; p.relu = 1; launch_gemm(p, 1, stream); }
  { GemmParams p = mkgp(S.tcat, 0, theo.ff.w, theo.ff.b, S.tmpD, 0, CB * CNT, CD, 2 * CD);
    p.relu = 1; launch_gemm(p, 1, stream); }
  k_theo_wsum<<<CB, 256, 0, stream>>>(TBT, S.wsumT, CET);
  run_graph_mha(stream, S, S.tmpD, CNT, TBI, CET, S.wsumT, 1, theo.ga, S.tA, CD, 1);

  // ---------------- encoder stack ----------------
  for (int l = 0; l < 4; ++l) {
    run_mha(stream, S, S.xA, S.xA, CNC, CNC, enc[l].mha, S.attnOut, CD, 0);
    k_ln_res<<<CB * CNC, 128, 0, stream>>>(S.attnOut, S.xA, enc[l].ln.g, enc[l].ln.b, S.xB);
    { GemmParams p = mkgp(S.xB, 0, enc[l].ff1.w, enc[l].ff1.b, S.ffTmp, 0, CB * CNC, CDFF, CD);
      p.relu = 1; launch_gemm(p, 1, stream); }
    { GemmParams p = mkgp(S.ffTmp, 0, enc[l].ff2.w, enc[l].ff2.b, S.tmpD, 0, CB * CNC, CD, CDFF);
      launch_gemm(p, 1, stream); }
    k_ln_res<<<CB * CNC, 128, 0, stream>>>(S.tmpD, S.xB, enc[l].ln.g, enc[l].ln.b, S.xA);
  }

  // ---------------- decoder stack ----------------
  for (int l = 0; l < 4; ++l) {
    run_mha(stream, S, S.tA, S.tA, CNT, CNT, dec[l].attn, S.attnOut, CD, 1);
    k_ln_res<<<CB * CNT, 128, 0, stream>>>(S.attnOut, S.tA, dec[l].ln.g, dec[l].ln.b, S.tB);
    run_mha(stream, S, S.tB, S.xA, CNT, CNC, dec[l].cross, S.attnOut, CD, 0);
    k_ln_res<<<CB * CNT, 128, 0, stream>>>(S.attnOut, S.tB, dec[l].ln.g, dec[l].ln.b, S.tC);
    { GemmParams p = mkgp(S.tC, 0, dec[l].ff1.w, dec[l].ff1.b, S.ffTmp, 0, CB * CNT, CDFF, CD);
      p.relu = 1; launch_gemm(p, 1, stream); }
    { GemmParams p = mkgp(S.ffTmp, 0, dec[l].ff2.w, dec[l].ff2.b, S.tmpD, 0, CB * CNT, CD, CDFF);
      launch_gemm(p, 1, stream); }
    k_ln_res<<<CB * CNT, 128, 0, stream>>>(S.tmpD, S.tC, dec[l].ln.g, dec[l].ln.b, S.tA);
  }

  // ---------------- transducer ----------------
  float* outElem  = (float*)d_out;                          // (B,NT,ELEM)
  float* outPos   = outElem + (size_t)CB * CNT * CELEM;     // (B,NT,3)
  float* outBonds = outPos + (size_t)CB * CNT * 3;          // (B,NT,NT,NB)

  // elem = softmax_axis0(e2(relu(e1(t))))
  { GemmParams p = mkgp(S.tA, 0, w_e1.w, w_e1.b, S.tmpD, 0, CB * CNT, CD, CD);
    p.relu = 1; launch_gemm(p, 1, stream); }
  { GemmParams p = mkgp(S.tmpD, 0, w_e2.w, w_e2.b, outElem, 0, CB * CNT, CELEM, CD);
    launch_gemm(p, 1, stream); }
  k_softmax_axis0<<<(CNT * CELEM + 255) / 256, 256, 0, stream>>>(outElem, CNT * CELEM);

  // pos = p2(relu(p1(t)))
  { GemmParams p = mkgp(S.tA, 0, w_p1.w, w_p1.b, S.tmpD, 0, CB * CNT, CD, CD);
    p.relu = 1; launch_gemm(p, 1, stream); }
  { GemmParams p = mkgp(S.tmpD, 0, w_p2.w, w_p2.b, outPos, 0, CB * CNT, 3, CD);
    launch_gemm(p, 1, stream); }

  // bonds: slab the giant u=(B*NT, NT*D) tensor in 128-row chunks, keep in f16 / L2
  for (int sb = 0; sb < (CB * CNT) / 128; ++sb) {
    const float* Aslab = S.tA + (size_t)sb * 128 * CD;
    { GemmParams p = mkgp(Aslab, 0, w_b1.w, w_b1.b, S.u16, 1, 128, CNT * CD, CD);
      p.relu = 1; launch_gemm(p, 1, stream); }
    { GemmParams p = mkgp(S.u16, 1, w_b2a.w, w_b2a.b, S.v16, 1, 128 * CNT, CD, CD);
      p.relu = 1; launch_gemm(p, 1, stream); }
    { GemmParams p = mkgp(S.v16, 1, w_b2b.w, w_b2b.b,
                          outBonds + (size_t)sb * 128 * CNT * CNB, 0, 128 * CNT, CNB, CD);
      launch_gemm(p, 1, stream); }
  }
  k_softmax_bonds<<<CB * CNT * CNB, 256, 0, stream>>>(outBonds);
}
